// SpeMamba_8718783611645
// MI455X (gfx1250) — compile-verified, matrix operand order checked by reading, not run on hardware
//
#include <hip/hip_runtime.h>
#include <stdint.h>

// ---------------------------------------------------------------------------
// SPE-Mamba fused kernel for MI455X (gfx1250), wave32 + WMMA bf16 + TDM.
//
// Pipeline:
//   K0: zero 32-float GroupNorm stats accumulator in d_ws
//   K1: fused  in_proj(WMMA) -> causal dwconv+SiLU (register-local over T)
//       -> x_proj(WMMA) -> softplus(dt_proj) -> selective scan (f32 VALU)
//       -> gate with SiLU(z) -> out_proj(WMMA) -> pre-GN output to d_out
//       (+ per-(batch,group) sum/sumsq partials via LDS/global f32 atomics)
//       x tile in / out tile back via Tensor Data Mover (2D tile descriptor).
//   K2: GroupNorm finalize + SiLU + residual, in place on d_out.
// ---------------------------------------------------------------------------

#if __has_builtin(__builtin_amdgcn_tensor_load_to_lds)
#define HAVE_TDM 1
#else
#define HAVE_TDM 0
#endif

typedef __attribute__((ext_vector_type(16))) __bf16 v16bf;
typedef __attribute__((ext_vector_type(8)))  float  v8f;
typedef __attribute__((ext_vector_type(4)))  unsigned int u32x4;
typedef __attribute__((ext_vector_type(8)))  int    i32x8;
typedef __attribute__((ext_vector_type(4)))  int    i32x4;

#define NB     4
#define NC     256
#define NH     64
#define NW     64
#define TOK    8
#define DMdim  32
#define DIdim  64
#define DSdim  16
#define NFRAG  18          // 8 in_proj + 6 x_proj + 4 out_proj B-fragments

union FragBF { v16bf v; uint32_t w[8]; uint4 q[2]; };

__device__ __forceinline__ float siluf(float v)     { return v / (1.0f + __expf(-v)); }
__device__ __forceinline__ float softplusf(float v) { return (v > 20.f) ? v : log1pf(__expf(v)); }
__device__ __forceinline__ uint32_t f2bf(float a) {  // f32 -> bf16 bits, RNE
  union { float f; uint32_t u; } c; c.f = a;
  return (c.u + 0x7FFFu + ((c.u >> 16) & 1u)) >> 16;
}

// B-fragment weight fetch: frag f, column n (0..15), K index k (0..31).
__device__ __forceinline__ float wval(int f, int n, int k,
                                      const float* __restrict__ ipw,
                                      const float* __restrict__ xpw,
                                      const float* __restrict__ opw) {
  if (f < 8)  { int e = f * 16 + n; return ipw[e * DMdim + k]; }                 // k < 32
  if (f < 14) { int t = f - 8; int e = (t >> 1) * 16 + n; int d = (t & 1) * 32 + k;
                return (e < 34) ? xpw[e * DIdim + d] : 0.f; }                    // pad 34->48
  { int t = f - 14; int e = (t >> 1) * 16 + n; int d = (t & 1) * 32 + k;
    return opw[e * DIdim + d]; }
}

#if HAVE_TDM
// 2D TDM tile: tile_dim0=16 pixels (X, contiguous in w), tile_dim1=256 channel
// rows at stride H*W elements; data_size=4B; groups 2/3 zero (<=2D tensor).
// D# field packing per CDNA5 ISA ch.8 (group0: count/lds_addr/global_addr/type,
// group1: data_size, tensor_dim0/1, tile_dim0/1, tensor_dim0_stride).
// This toolchain exposes the 6-arg builtin form:
//   (u32x4 g0, i32x8 g1, i32x4 g2, i32x4 g3, i32x8 g4, i32 cpol)
__device__ __forceinline__ void tdm_tile_2d(bool store, const float* gptr, uint32_t ldsoff) {
  u32x4 g0; i32x8 g1;
  i32x4 gz4; gz4[0] = gz4[1] = gz4[2] = gz4[3] = 0;
  i32x8 gz8; gz8[0] = gz8[1] = gz8[2] = gz8[3] = 0;
             gz8[4] = gz8[5] = gz8[6] = gz8[7] = 0;
  uint64_t ga = (uint64_t)(uintptr_t)gptr;
  g0[0] = 1u;                                                    // count=1, user mode
  g0[1] = ldsoff;                                                // lds_addr (bytes)
  g0[2] = (uint32_t)ga;                                          // global_addr[31:0]
  g0[3] = (uint32_t)((ga >> 32) & 0x1FFFFFFull) | (2u << 30);    // addr[56:32] | type=2
  g1[0] = (int)(2u << 16);                                       // data_size=4B
  g1[1] = (int)((uint32_t)NW << 16);                             // tensor_dim0=64
  g1[2] = (int)((uint32_t)NC << 16);                             // tensor_dim1=256
  g1[3] = (int)(16u << 16);                                      // tile_dim0=16
  g1[4] = NC;                                                    // tile_dim1=256, tile_dim2=0
  g1[5] = NH * NW;                                               // tensor_dim0_stride=4096
  g1[6] = 0;
  g1[7] = 0;
  if (store) __builtin_amdgcn_tensor_store_from_lds(g0, g1, gz4, gz4, gz8, 0);
  else       __builtin_amdgcn_tensor_load_to_lds (g0, g1, gz4, gz4, gz8, 0);
}
#endif

__global__ void zero_stats_kernel(float* g) {
  if (threadIdx.x < 32) g[threadIdx.x] = 0.f;
}

// ---------------------------------------------------------------------------
// K1: fused mamba. Grid: 1024 WGs x 256 threads (8 waves). WG = (b, h, 16 w's).
// Wave mt owns tokens 16mt..16mt+15 = pixels {2mt, 2mt+1} x T=8.
// WMMA layouts (ISA 7.12.2, wave32):
//   A(16xK bf16): lane L -> row L&15; 16 elems: K = (L>>4)*8 + (i<8 ? i : i+8)
//   B(Kx16 bf16): lane L -> col L&15; same K striping (B column-major == A^T)
//   C/D(16x16 f32): reg r, lane L -> row r + 8*(L>>4), col L&15
// ---------------------------------------------------------------------------
__global__ __launch_bounds__(256) void mamba_fused_kernel(
    const float* __restrict__ x,   const float* __restrict__ ipw,
    const float* __restrict__ cw,  const float* __restrict__ cbv,
    const float* __restrict__ xpw, const float* __restrict__ dtw,
    const float* __restrict__ dtb, const float* __restrict__ alog,
    const float* __restrict__ Dpv, const float* __restrict__ opw,
    float* __restrict__ outbuf, float* gstats)
{
  __shared__ float    xt[NC * 16];                 // x tile  [c][p]     16 KB
  __shared__ uint32_t bfrag[NFRAG * 32 * 8];       // bf16 B fragments   18 KB
  __shared__ float    u2buf[8][16 * DIdim];        // per-wave u2 / y    32 KB
  __shared__ float    szbuf[8][16 * DIdim];        // per-wave silu(z)   32 KB
  __shared__ float    xdbl[8][16 * 36];            // per-wave dt|B|C    18 KB
  __shared__ float    outLDS[NC * 16];             // out tile [c][p]    16 KB
  __shared__ float    lstats[8];                   // per-group sum/sumsq

  const int tid  = threadIdx.x;
  const int lane = tid & 31;
  const int wv   = tid >> 5;           // wave id == M-tile id (0..7)
  const int g    = blockIdx.x;
  const int wblk = g & 3, hh = (g >> 2) & 63, bb = g >> 8;
  const long long gbase = ((long long)bb * NC * NH + hh) * NW + wblk * 16; // c=0 elem

  if (tid < 8) lstats[tid] = 0.f;

#if HAVE_TDM
  if (tid == 0) {                       // one TDM load per WG (TDM ignores EXEC)
    tdm_tile_2d(false, x + gbase, (uint32_t)(uintptr_t)&xt[0]);
    __builtin_amdgcn_s_wait_tensorcnt(0);
  }
#else
  for (int i = tid; i < NC * 16; i += 256) {
    int c = i >> 4, p = i & 15;
    xt[i] = x[gbase + (long long)c * (NH * NW) + p];
  }
#endif

  // Cooperative pre-swizzle of all GEMM weights into bf16 B fragments.
  for (int slot = tid; slot < NFRAG * 32; slot += 256) {
    int f = slot >> 5, ln = slot & 31;
    int n = ln & 15, kbase = (ln >> 4) * 8;
    uint32_t* dst = &bfrag[slot * 8];
#pragma unroll
    for (int pr = 0; pr < 8; ++pr) {
      int i0 = 2 * pr, i1 = i0 + 1;
      int k0 = kbase + (i0 < 8 ? i0 : i0 + 8);
      int k1 = kbase + (i1 < 8 ? i1 : i1 + 8);
      dst[pr] = f2bf(wval(f, n, k0, ipw, xpw, opw)) |
                (f2bf(wval(f, n, k1, ipw, xpw, opw)) << 16);
    }
  }
  __syncthreads();

  const int mt    = wv;
  const int rA    = lane & 15;          // A row = token in tile
  const int kb    = (lane >> 4) * 8;    // K base for this half-wave
  const int pA    = 2 * mt + (rA >> 3); // pixel of token rA
  const int tA    = rA & 7;             // timestep of token rA
  const int col   = lane & 15;          // D column
  const int rhalf = lane >> 4;          // which 8 D rows this lane holds

  // ---- in_proj: A = x tokens (K=DM=32), 8 N-tiles of 128 outputs ----------
  FragBF a_in;
#pragma unroll
  for (int pr = 0; pr < 8; ++pr) {
    int i0 = 2 * pr, i1 = i0 + 1;
    int k0 = kb + (i0 < 8 ? i0 : i0 + 8);
    int k1 = kb + (i1 < 8 ? i1 : i1 + 8);
    a_in.w[pr] = f2bf(xt[(tA * DMdim + k0) * 16 + pA]) |
                 (f2bf(xt[(tA * DMdim + k1) * 16 + pA]) << 16);
  }

#pragma unroll
  for (int nt = 0; nt < 8; ++nt) {
    FragBF b;
    const uint4* bp = (const uint4*)&bfrag[(nt * 32 + lane) * 8];
    b.q[0] = bp[0]; b.q[1] = bp[1];
    v8f acc = {};
    acc = __builtin_amdgcn_wmma_f32_16x16x32_bf16(false, a_in.v, false, b.v,
                                                  (short)0, acc, false, false);
    if (nt < 4) {
      // u path: causal depthwise conv over T lives inside the accumulator regs
      int d = nt * 16 + col;
      float w0 = cw[d * 4 + 0], w1 = cw[d * 4 + 1];
      float w2 = cw[d * 4 + 2], w3 = cw[d * 4 + 3], bc = cbv[d];
      float u[8];
#pragma unroll
      for (int rr = 0; rr < 8; ++rr) u[rr] = acc[rr];
#pragma unroll
      for (int rr = 0; rr < 8; ++rr) {
        float s = bc + w3 * u[rr];
        if (rr >= 1) s += w2 * u[rr - 1];
        if (rr >= 2) s += w1 * u[rr - 2];
        if (rr >= 3) s += w0 * u[rr - 3];
        u2buf[wv][(rr + 8 * rhalf) * DIdim + d] = siluf(s);
      }
    } else {
      int d = (nt - 4) * 16 + col;
#pragma unroll
      for (int rr = 0; rr < 8; ++rr)
        szbuf[wv][(rr + 8 * rhalf) * DIdim + d] = siluf(acc[rr]);
    }
  }
  __syncthreads();   // cross-lane LDS handoff (D-layout -> A-layout)

  // ---- x_proj: A = u2 (K=DI=64 -> 2 chained WMMAs), 34 outputs (3 tiles) --
  FragBF a0, a1;
#pragma unroll
  for (int pr = 0; pr < 8; ++pr) {
    int i0 = 2 * pr, i1 = i0 + 1;
    int k0 = kb + (i0 < 8 ? i0 : i0 + 8);
    int k1 = kb + (i1 < 8 ? i1 : i1 + 8);
    a0.w[pr] = f2bf(u2buf[wv][rA * DIdim + k0]) |
               (f2bf(u2buf[wv][rA * DIdim + k1]) << 16);
    a1.w[pr] = f2bf(u2buf[wv][rA * DIdim + 32 + k0]) |
               (f2bf(u2buf[wv][rA * DIdim + 32 + k1]) << 16);
  }
#pragma unroll
  for (int nt = 0; nt < 3; ++nt) {
    FragBF b0, b1;
    const uint4* p0 = (const uint4*)&bfrag[((8 + nt * 2) * 32 + lane) * 8];
    const uint4* p1 = (const uint4*)&bfrag[((9 + nt * 2) * 32 + lane) * 8];
    b0.q[0] = p0[0]; b0.q[1] = p0[1];
    b1.q[0] = p1[0]; b1.q[1] = p1[1];
    v8f acc = {};
    acc = __builtin_amdgcn_wmma_f32_16x16x32_bf16(false, a0.v, false, b0.v,
                                                  (short)0, acc, false, false);
    acc = __builtin_amdgcn_wmma_f32_16x16x32_bf16(false, a1.v, false, b1.v,
                                                  (short)0, acc, false, false);
    int e = nt * 16 + col;
    if (e < 34) {
#pragma unroll
      for (int rr = 0; rr < 8; ++rr)
        xdbl[wv][(rr + 8 * rhalf) * 36 + e] = acc[rr];
    }
  }
  __syncthreads();

  // ---- selective scan (f32 VALU): lane = (pixel, 4 d-channels) ------------
  {
    const int pl = lane >> 4;            // pixel within wave
    const int dbase = (lane & 15) * 4;   // 4 consecutive d per lane
    float h[4][16];
    float An[4][16];
    float dw0[4], dw1[4], dbv[4], Dq[4];
#pragma unroll
    for (int j = 0; j < 4; ++j) {
      int d = dbase + j;
      dw0[j] = dtw[d * 2 + 0]; dw1[j] = dtw[d * 2 + 1];
      dbv[j] = dtb[d];         Dq[j]  = Dpv[d];
#pragma unroll
      for (int s = 0; s < 16; ++s) {
        h[j][s]  = 0.f;
        An[j][s] = -__expf(alog[d * DSdim + s]);   // A = -exp(A_log)
      }
    }
#pragma unroll
    for (int t = 0; t < TOK; ++t) {
      int r = pl * 8 + t;
      float d0 = xdbl[wv][r * 36 + 0], d1 = xdbl[wv][r * 36 + 1];
      float Bv[16], Cv[16];
#pragma unroll
      for (int s = 0; s < 16; ++s) {
        Bv[s] = xdbl[wv][r * 36 + 2 + s];
        Cv[s] = xdbl[wv][r * 36 + 18 + s];
      }
#pragma unroll
      for (int j = 0; j < 4; ++j) {
        int d = dbase + j;
        float delta = softplusf(d0 * dw0[j] + d1 * dw1[j] + dbv[j]);
        float uu = u2buf[wv][r * DIdim + d];
        float du = delta * uu;
        float y = 0.f;
#pragma unroll
        for (int s = 0; s < 16; ++s) {
          float dA = __expf(delta * An[j][s]);
          h[j][s] = h[j][s] * dA + du * Bv[s];
          y += h[j][s] * Cv[s];
        }
        y = (y + uu * Dq[j]) * szbuf[wv][r * DIdim + d];
        u2buf[wv][r * DIdim + d] = y;   // ybuf aliases u2buf (exclusive slot)
      }
    }
  }
  __syncthreads();

  // ---- out_proj: A = y (K=64), 32 outputs (2 tiles) -----------------------
  FragBF y0, y1;
#pragma unroll
  for (int pr = 0; pr < 8; ++pr) {
    int i0 = 2 * pr, i1 = i0 + 1;
    int k0 = kb + (i0 < 8 ? i0 : i0 + 8);
    int k1 = kb + (i1 < 8 ? i1 : i1 + 8);
    y0.w[pr] = f2bf(u2buf[wv][rA * DIdim + k0]) |
               (f2bf(u2buf[wv][rA * DIdim + k1]) << 16);
    y1.w[pr] = f2bf(u2buf[wv][rA * DIdim + 32 + k0]) |
               (f2bf(u2buf[wv][rA * DIdim + 32 + k1]) << 16);
  }
#pragma unroll
  for (int nt = 0; nt < 2; ++nt) {
    FragBF b0, b1;
    const uint4* p0 = (const uint4*)&bfrag[((14 + nt * 2) * 32 + lane) * 8];
    const uint4* p1 = (const uint4*)&bfrag[((15 + nt * 2) * 32 + lane) * 8];
    b0.q[0] = p0[0]; b0.q[1] = p0[1];
    b1.q[0] = p1[0]; b1.q[1] = p1[1];
    v8f acc = {};
    acc = __builtin_amdgcn_wmma_f32_16x16x32_bf16(false, y0.v, false, b0.v,
                                                  (short)0, acc, false, false);
    acc = __builtin_amdgcn_wmma_f32_16x16x32_bf16(false, y1.v, false, b1.v,
                                                  (short)0, acc, false, false);
    int m = nt * 16 + col;
    int p = 2 * mt + rhalf;
#pragma unroll
    for (int rr = 0; rr < 8; ++rr)                 // token (p, t=rr), channel c
      outLDS[(rr * DMdim + m) * 16 + p] = acc[rr];
  }
  __syncthreads();

  // ---- GroupNorm partials + output tile store -----------------------------
  {
    int c = tid;
    float s = 0.f, ss = 0.f;
#pragma unroll
    for (int p = 0; p < 16; ++p) {
      float v = outLDS[c * 16 + p];
      s += v; ss += v * v;
    }
    atomicAdd(&lstats[(c >> 6) * 2 + 0], s);       // ds_add_f32
    atomicAdd(&lstats[(c >> 6) * 2 + 1], ss);
  }
  __syncthreads();
  if (tid < 8)
    atomicAdd(&gstats[(bb * 4 + (tid >> 1)) * 2 + (tid & 1)], lstats[tid]);

#if HAVE_TDM
  if (tid == 0) {
    tdm_tile_2d(true, outbuf + gbase, (uint32_t)(uintptr_t)&outLDS[0]);
    __builtin_amdgcn_s_wait_tensorcnt(0);
  }
#else
  for (int i = tid; i < NC * 16; i += 256) {
    int c = i >> 4, p = i & 15;
    outbuf[gbase + (long long)c * (NH * NW) + p] = outLDS[i];
  }
#endif
}

// ---------------------------------------------------------------------------
// K2: GroupNorm finalize + SiLU + residual, in place on d_out (float4 path).
// ---------------------------------------------------------------------------
__global__ __launch_bounds__(256) void gn_finalize_kernel(
    const float* __restrict__ x, const float* __restrict__ gnw,
    const float* __restrict__ gnb, const float* __restrict__ gstats,
    float* out)
{
  const int total4 = NB * NC * NH * NW / 4;
  int q = blockIdx.x * blockDim.x + threadIdx.x;
  if (q >= total4) return;
  int e = q * 4;
  int c = (e >> 12) & (NC - 1);
  int b = e >> 20;
  int grp = c >> 6;
  const float inv = 1.0f / (64.0f * NH * NW);
  float sum = gstats[(b * 4 + grp) * 2 + 0];
  float ssq = gstats[(b * 4 + grp) * 2 + 1];
  float mu  = sum * inv;
  float var = ssq * inv - mu * mu;
  float rstd = rsqrtf(var + 1e-5f);
  float wsc = gnw[c] * rstd, bia = gnb[c] - mu * wsc;
  float4 o  = ((const float4*)out)[q];
  float4 xv = ((const float4*)x)[q];
  float4 r;
  {
    float xn;
    xn = o.x * wsc + bia; r.x = xv.x + xn / (1.f + __expf(-xn));
    xn = o.y * wsc + bia; r.y = xv.y + xn / (1.f + __expf(-xn));
    xn = o.z * wsc + bia; r.z = xv.z + xn / (1.f + __expf(-xn));
    xn = o.w * wsc + bia; r.w = xv.w + xn / (1.f + __expf(-xn));
  }
  ((float4*)out)[q] = r;
}

extern "C" void kernel_launch(void* const* d_in, const int* in_sizes, int n_in,
                              void* d_out, int out_size, void* d_ws, size_t ws_size,
                              hipStream_t stream) {
  (void)in_sizes; (void)n_in; (void)out_size; (void)ws_size;
  const float* x    = (const float*)d_in[0];
  const float* ipw  = (const float*)d_in[1];
  const float* cw   = (const float*)d_in[2];
  const float* cbv  = (const float*)d_in[3];
  const float* xpw  = (const float*)d_in[4];
  const float* dtw  = (const float*)d_in[5];
  const float* dtb  = (const float*)d_in[6];
  const float* alog = (const float*)d_in[7];
  const float* Dpv  = (const float*)d_in[8];
  const float* opw  = (const float*)d_in[9];
  const float* gnw  = (const float*)d_in[10];
  const float* gnb  = (const float*)d_in[11];
  float* out    = (float*)d_out;     // pre-GN intermediate lives here too
  float* gstats = (float*)d_ws;      // 32 floats: (b,g) x {sum, sumsq}

  zero_stats_kernel<<<1, 32, 0, stream>>>(gstats);

  const int nblocks = (NB * NH * NW) / 16;   // 1024 WGs of 16 pixels
  mamba_fused_kernel<<<nblocks, 256, 0, stream>>>(
      x, ipw, cw, cbv, xpw, dtw, dtb, alog, Dpv, opw, out, gstats);

  const int total4 = NB * NC * NH * NW / 4;
  gn_finalize_kernel<<<(total4 + 255) / 256, 256, 0, stream>>>(
      x, gnw, gnb, gstats, out);
}
